// SelfAttention_47991964566247
// MI455X (gfx1250) — matrix operand order repaired and checked by reading, not compiled, not run on hardware
//
#include <hip/hip_runtime.h>
#include <hip/hip_bf16.h>
#include <stdint.h>

// ---------------------------------------------------------------------------
// Types for gfx1250 WMMA (wave32): A/B = 16 bf16 per lane, C/D = 8 f32 per lane
// ---------------------------------------------------------------------------
typedef __attribute__((ext_vector_type(8)))  unsigned short u16x8;
typedef __attribute__((ext_vector_type(16))) unsigned short u16x16;
typedef __attribute__((ext_vector_type(16))) __bf16         bf16x16;
typedef __attribute__((ext_vector_type(8)))  float          f32x8;

#define N_TOK 4096
#define DIM   768
#define HEADS 12
#define HD    64
#define ATT_SCALE 0.125f   // 64^-0.5

// Hardware f32 -> bf16 convert (backend emits v_cvt_pk_bf16_f32).
static __device__ __forceinline__ unsigned short f2bf(float f) {
    return __builtin_bit_cast(unsigned short, static_cast<__bf16>(f));
}

// ---------------------------------------------------------------------------
// Async global -> LDS copy (GLOBAL_LOAD_ASYNC_TO_LDS_B128, ASYNCcnt-tracked).
// Each active lane moves 16 bytes; LDS dest address is per-lane (VDST VGPR),
// global source is a per-lane 64-bit address. Low 32 bits of a generic pointer
// to a __shared__ object are the wave-relative LDS byte address.
// ---------------------------------------------------------------------------
static __device__ __forceinline__ void async_copy_b128(void* lds_dst, const void* gsrc) {
    unsigned lds_off = (unsigned)(size_t)lds_dst;
    asm volatile("global_load_async_to_lds_b128 %0, %1, off"
                 :: "v"(lds_off), "v"(gsrc) : "memory");
}
static __device__ __forceinline__ void wait_async() {
    asm volatile("s_wait_asynccnt 0x0" ::: "memory");
}

// A-matrix fragment (16x32, MxK, bf16). Caller passes pointer to
// lds + (row_base + (lane&15))*stride + k_off + (lane>>4)*16.
// Elements 0..15 are 16 contiguous bf16 (k = 16*(lane>>4) + s).
static __device__ __forceinline__ bf16x16 frag_a(const unsigned short* p) {
    u16x8 lo = *(const u16x8*)(p);
    u16x8 hi = *(const u16x8*)(p + 8);
    u16x16 u = __builtin_shufflevector(lo, hi, 0,1,2,3,4,5,6,7,8,9,10,11,12,13,14,15);
    return __builtin_bit_cast(bf16x16, u);
}

// B-matrix fragment (32x16, KxN, bf16) from a col-major LDS tile [col][k].
// Caller passes lds + (col_base + (lane&15))*stride + k_off + (lane>>4)*16.
static __device__ __forceinline__ bf16x16 frag_b(const unsigned short* p) {
    u16x8 lo = *(const u16x8*)(p);
    u16x8 hi = *(const u16x8*)(p + 8);
    u16x16 u = __builtin_shufflevector(lo, hi, 0,1,2,3,4,5,6,7,8,9,10,11,12,13,14,15);
    return __builtin_bit_cast(bf16x16, u);
}

static __device__ __forceinline__ f32x8 wmma_bf16(bf16x16 a, bf16x16 b, f32x8 c) {
    return __builtin_amdgcn_wmma_f32_16x16x32_bf16(
        /*neg_a=*/false, a, /*neg_b=*/false, b,
        /*c_mod=*/(short)0, c, /*reuse_a=*/false, /*reuse_b=*/false);
}

// ---------------------------------------------------------------------------
// Kernel 1: QKV = X[4096,768] @ W_qkv[768,2304], bf16 WMMA, f32 accumulate.
// 256 thr (8 waves), tile 128x64, BK=32; each wave owns a 32x32 quadrant.
// Fills need f32->bf16 conversion, so they use the VGPR (cvt_pk) path.
// Output tile is always inside one of Q/K/V and one head, so routing is SALU.
// ---------------------------------------------------------------------------
__global__ __launch_bounds__(256) void qkv_gemm_kernel(
    const float* __restrict__ X, const float* __restrict__ W,
    unsigned short* __restrict__ Q, unsigned short* __restrict__ K,
    unsigned short* __restrict__ V)
{
    __shared__ unsigned short ldsA[128 * 32];  // [row][k]
    __shared__ unsigned short ldsB[64 * 32];   // [col][k]
    const int tid  = threadIdx.x;
    const int lane = tid & 31, wave = tid >> 5;
    const int m0 = blockIdx.x * 128;
    const int c0 = blockIdx.y * 64;
    const int mw = (wave >> 1) * 32, nw = (wave & 1) * 32;
    const int lr = lane & 15, hi = lane >> 4;

    const int which = c0 / DIM;               // 0=Q 1=K 2=V (block-uniform)
    const int h     = (c0 % DIM) / HD;
    unsigned short* __restrict__ base = (which == 0) ? Q : ((which == 1) ? K : V);
    const float oscale = (which == 0) ? ATT_SCALE : 1.0f;

    f32x8 acc[2][2] = {};

    for (int k0 = 0; k0 < DIM; k0 += 32) {
        __syncthreads();
        { // A tile 128x32 f32 -> bf16, row-major
            int r = tid >> 3, c4 = (tid & 7) * 4;
            #pragma unroll
            for (int p = 0; p < 4; ++p) {
                int row = p * 32 + r;
                const float4 v = *(const float4*)(X + (size_t)(m0 + row) * DIM + k0 + c4);
                unsigned short* d = ldsA + row * 32 + c4;
                d[0] = f2bf(v.x); d[1] = f2bf(v.y); d[2] = f2bf(v.z); d[3] = f2bf(v.w);
            }
        }
        { // B tile 32x64 f32 -> bf16, stored col-major [col][k]
            int c4 = (tid & 15) * 4, kr = tid >> 4;
            #pragma unroll
            for (int p = 0; p < 2; ++p) {
                int kk = p * 16 + kr;
                const float4 v = *(const float4*)(W + (size_t)(k0 + kk) * (3 * DIM) + c0 + c4);
                ldsB[(c4 + 0) * 32 + kk] = f2bf(v.x);
                ldsB[(c4 + 1) * 32 + kk] = f2bf(v.y);
                ldsB[(c4 + 2) * 32 + kk] = f2bf(v.z);
                ldsB[(c4 + 3) * 32 + kk] = f2bf(v.w);
            }
        }
        // prefetch next k-step while this one computes (global_prefetch_b8)
        if (k0 + 32 < DIM) {
            __builtin_prefetch(X + (size_t)(m0 + (tid >> 1)) * DIM + k0 + 32 + (tid & 1) * 16, 0, 1);
            __builtin_prefetch(W + (size_t)(k0 + 32 + (tid >> 4)) * (3 * DIM) + c0 + (tid & 15) * 4, 0, 1);
        }
        __syncthreads();

        bf16x16 a0 = frag_a(ldsA + (mw +      lr) * 32 + hi * 16);
        bf16x16 a1 = frag_a(ldsA + (mw + 16 + lr) * 32 + hi * 16);
        bf16x16 b0 = frag_b(ldsB + (nw +      lr) * 32 + hi * 16);
        bf16x16 b1 = frag_b(ldsB + (nw + 16 + lr) * 32 + hi * 16);
        acc[0][0] = wmma_bf16(a0, b0, acc[0][0]);
        acc[0][1] = wmma_bf16(a0, b1, acc[0][1]);
        acc[1][0] = wmma_bf16(a1, b0, acc[1][0]);
        acc[1][1] = wmma_bf16(a1, b1, acc[1][1]);
    }

    const size_t hb = (size_t)h * N_TOK;
    #pragma unroll
    for (int ar = 0; ar < 2; ++ar)
    #pragma unroll
    for (int ac = 0; ac < 2; ++ac)
    #pragma unroll
    for (int r = 0; r < 8; ++r) {
        int row = m0 + mw + ar * 16 + r + hi * 8;
        int hd  = nw + ac * 16 + lr;
        base[(hb + row) * HD + hd] = f2bf(acc[ar][ac][r] * oscale);
    }
}

// ---------------------------------------------------------------------------
// Kernel 2: flash attention per (head, 64-query block). 4 waves, each owns
// 16 query rows. Q/K tiles stream through async global->LDS DMA (ASYNCcnt);
// V is transposed in flight so it keeps the VGPR path. Online softmax; S and
// O live in WMMA accumulators.
// ---------------------------------------------------------------------------
__global__ __launch_bounds__(128) void flash_attn_kernel(
    const unsigned short* __restrict__ Q, const unsigned short* __restrict__ K,
    const unsigned short* __restrict__ V, unsigned short* __restrict__ O)
{
    __shared__ unsigned short ldsQ[64 * 64];     // [row][hd]
    __shared__ unsigned short ldsK[64 * 64];     // [key][hd]  (B-frag col-major)
    __shared__ unsigned short ldsV[64 * 64];     // [hd][key]  (B-frag col-major)
    __shared__ unsigned short ldsP[4 * 16 * 64]; // per-wave P tile [row][key]

    const int tid = threadIdx.x, lane = tid & 31, wave = tid >> 5;
    const int lr = lane & 15, hi = lane >> 4;
    const int h  = blockIdx.y;
    const int q0 = blockIdx.x * 64;
    const size_t hb = (size_t)h * N_TOK * HD;

    #pragma unroll
    for (int p = 0; p < 4; ++p) {   // Q block: async DMA, no VGPR staging
        int idx = p * 128 + tid;
        int row = idx >> 3, c8 = (idx & 7) * 8;
        async_copy_b128(ldsQ + row * 64 + c8,
                        Q + hb + (size_t)(q0 + row) * HD + c8);
    }

    f32x8 o[4] = {};
    float m_s[8], l_s[8];
    #pragma unroll
    for (int r = 0; r < 8; ++r) { m_s[r] = -1e30f; l_s[r] = 0.f; }

    for (int kb = 0; kb < N_TOK; kb += 64) {
        __syncthreads();            // prior-iteration readers done
        #pragma unroll
        for (int p = 0; p < 4; ++p) {   // K tile: async DMA
            int idx = p * 128 + tid;
            int row = idx >> 3, c8 = (idx & 7) * 8;
            async_copy_b128(ldsK + row * 64 + c8,
                            K + hb + (size_t)(kb + row) * HD + c8);
        }
        #pragma unroll
        for (int p = 0; p < 4; ++p) {   // V tile, transposed to [hd][key]
            int key = tid & 63;
            int chunk = (tid >> 6) + p * 2;
            u16x8 v = *(const u16x8*)(V + hb + (size_t)(kb + key) * HD + chunk * 8);
            #pragma unroll
            for (int j = 0; j < 8; ++j) ldsV[(chunk * 8 + j) * 64 + key] = v[j];
        }
        // prefetch next key-block (overlaps with WMMA below)
        if (kb + 64 < N_TOK) {
            __builtin_prefetch(K + hb + (size_t)(kb + 64 + (tid >> 1)) * HD + (tid & 1) * 32, 0, 1);
            __builtin_prefetch(V + hb + (size_t)(kb + 64 + (tid >> 1)) * HD + (tid & 1) * 32, 0, 1);
        }
        wait_async();               // our async LDS writes landed
        __syncthreads();            // everyone's writes visible

        // S = Q * K^T  (16 rows x 64 keys per wave)
        f32x8 s[4] = {};
        #pragma unroll
        for (int ks = 0; ks < 2; ++ks) {
            bf16x16 a = frag_a(ldsQ + (wave * 16 + lr) * 64 + ks * 32 + hi * 16);
            #pragma unroll
            for (int t = 0; t < 4; ++t) {
                bf16x16 b = frag_b(ldsK + (t * 16 + lr) * 64 + ks * 32 + hi * 16);
                s[t] = wmma_bf16(a, b, s[t]);
            }
        }

        // online softmax. C-layout: VGPR r <-> row r (lanes 0-15) / row r+8 (16-31);
        // shfl offsets <=8 keep the two halves independent.
        unsigned short* pw = ldsP + wave * (16 * 64);
        #pragma unroll
        for (int r = 0; r < 8; ++r) {
            float mt = fmaxf(fmaxf(s[0][r], s[1][r]), fmaxf(s[2][r], s[3][r]));
            #pragma unroll
            for (int off = 8; off >= 1; off >>= 1)
                mt = fmaxf(mt, __shfl_xor(mt, off, 32));
            float mn   = fmaxf(m_s[r], mt);
            float corr = __expf(m_s[r] - mn);
            l_s[r] *= corr;
            o[0][r] *= corr; o[1][r] *= corr; o[2][r] *= corr; o[3][r] *= corr;
            float rs = 0.f;
            #pragma unroll
            for (int t = 0; t < 4; ++t) {
                float pv = __expf(s[t][r] - mn);
                rs += pv;
                pw[(r + hi * 8) * 64 + t * 16 + lr] = f2bf(pv);
            }
            #pragma unroll
            for (int off = 8; off >= 1; off >>= 1)
                rs += __shfl_xor(rs, off, 32);
            l_s[r] += rs;
            m_s[r]  = mn;
        }

        // O += P * V   (wave-private ldsP; DScnt dependency handled by compiler)
        #pragma unroll
        for (int ks = 0; ks < 2; ++ks) {
            bf16x16 a = frag_a(pw + lr * 64 + ks * 32 + hi * 16);
            #pragma unroll
            for (int t = 0; t < 4; ++t) {
                bf16x16 b = frag_b(ldsV + (t * 16 + lr) * 64 + ks * 32 + hi * 16);
                o[t] = wmma_bf16(a, b, o[t]);
            }
        }
    }

    #pragma unroll
    for (int r = 0; r < 8; ++r) {
        float inv = 1.0f / l_s[r];
        int row = q0 + wave * 16 + r + hi * 8;
        #pragma unroll
        for (int t = 0; t < 4; ++t)
            O[(size_t)row * DIM + h * HD + t * 16 + lr] = f2bf(o[t][r] * inv);
    }
}

// ---------------------------------------------------------------------------
// Kernel 3: out = attn[4096,768](bf16) @ W_proj[768,768] + b_proj, f32 out.
// 256 thr, tile 128x64, BK=32. A tile is a straight bf16 copy -> async DMA.
// ---------------------------------------------------------------------------
__global__ __launch_bounds__(256) void proj_gemm_kernel(
    const unsigned short* __restrict__ A, const float* __restrict__ W,
    const float* __restrict__ bias, float* __restrict__ out)
{
    __shared__ unsigned short ldsA[128 * 32];
    __shared__ unsigned short ldsB[64 * 32];
    const int tid  = threadIdx.x;
    const int lane = tid & 31, wave = tid >> 5;
    const int m0 = blockIdx.x * 128;
    const int c0 = blockIdx.y * 64;
    const int mw = (wave >> 1) * 32, nw = (wave & 1) * 32;
    const int lr = lane & 15, hi = lane >> 4;

    f32x8 acc[2][2] = {};

    for (int k0 = 0; k0 < DIM; k0 += 32) {
        __syncthreads();
        { // A tile (128x32 bf16): async global->LDS DMA
            int r = tid >> 2, c8 = (tid & 3) * 8;
            #pragma unroll
            for (int p = 0; p < 2; ++p) {
                int row = p * 64 + r;
                async_copy_b128(ldsA + row * 32 + c8,
                                A + (size_t)(m0 + row) * DIM + k0 + c8);
            }
        }
        { // B tile f32 -> bf16, col-major [col][k]
            int c4 = (tid & 15) * 4, kr = tid >> 4;
            #pragma unroll
            for (int p = 0; p < 2; ++p) {
                int kk = p * 16 + kr;
                const float4 v = *(const float4*)(W + (size_t)(k0 + kk) * DIM + c0 + c4);
                ldsB[(c4 + 0) * 32 + kk] = f2bf(v.x);
                ldsB[(c4 + 1) * 32 + kk] = f2bf(v.y);
                ldsB[(c4 + 2) * 32 + kk] = f2bf(v.z);
                ldsB[(c4 + 3) * 32 + kk] = f2bf(v.w);
            }
        }
        if (k0 + 32 < DIM) {
            __builtin_prefetch(A + (size_t)(m0 + (tid >> 1)) * DIM + k0 + 32 + (tid & 1) * 16, 0, 1);
            __builtin_prefetch(W + (size_t)(k0 + 32 + (tid >> 4)) * DIM + c0 + (tid & 15) * 4, 0, 1);
        }
        wait_async();
        __syncthreads();

        bf16x16 a0 = frag_a(ldsA + (mw +      lr) * 32 + hi * 16);
        bf16x16 a1 = frag_a(ldsA + (mw + 16 + lr) * 32 + hi * 16);
        bf16x16 b0 = frag_b(ldsB + (nw +      lr) * 32 + hi * 16);
        bf16x16 b1 = frag_b(ldsB + (nw + 16 + lr) * 32 + hi * 16);
        acc[0][0] = wmma_bf16(a0, b0, acc[0][0]);
        acc[0][1] = wmma_bf16(a0, b1, acc[0][1]);
        acc[1][0] = wmma_bf16(a1, b0, acc[1][0]);
        acc[1][1] = wmma_bf16(a1, b1, acc[1][1]);
    }

    #pragma unroll
    for (int ar = 0; ar < 2; ++ar)
    #pragma unroll
    for (int ac = 0; ac < 2; ++ac)
    #pragma unroll
    for (int r = 0; r < 8; ++r) {
        int row = m0 + mw + ar * 16 + r + hi * 8;
        int col = c0 + nw + ac * 16 + lr;
        out[(size_t)row * DIM + col] = acc[ar][ac][r] + bias[col];
    }
}

// ---------------------------------------------------------------------------
extern "C" void kernel_launch(void* const* d_in, const int* in_sizes, int n_in,
                              void* d_out, int out_size, void* d_ws, size_t ws_size,
                              hipStream_t stream) {
    const float* x      = (const float*)d_in[0];
    const float* W_qkv  = (const float*)d_in[1];
    const float* W_proj = (const float*)d_in[2];
    const float* b_proj = (const float*)d_in[3];
    float* outf = (float*)d_out;

    // workspace layout (bf16): Q | K | V  [12][4096][64]  + attn [4096][768]
    const size_t HSZ = (size_t)HEADS * N_TOK * HD;           // 3,145,728 elems
    unsigned short* ws   = (unsigned short*)d_ws;
    unsigned short* qb   = ws;
    unsigned short* kb   = ws + HSZ;
    unsigned short* vb   = ws + 2 * HSZ;
    unsigned short* attn = ws + 3 * HSZ;

    qkv_gemm_kernel<<<dim3(N_TOK / 128, (3 * DIM) / 64), 256, 0, stream>>>(
        x, W_qkv, qb, kb, vb);
    flash_attn_kernel<<<dim3(N_TOK / 64, HEADS), 128, 0, stream>>>(
        qb, kb, vb, attn);
    proj_gemm_kernel<<<dim3(N_TOK / 128, DIM / 64), 256, 0, stream>>>(
        attn, W_proj, b_proj, outf);
}
